// IncConvModule_74028056314079
// MI455X (gfx1250) — matrix operand order repaired and checked by reading, not compiled, not run on hardware
//
#include <hip/hip_runtime.h>

// CDNA5 / gfx1250: wave32, WMMA 16x16x32 bf16, fp32 accumulate.
typedef __attribute__((ext_vector_type(16))) __bf16         v16bf;
typedef __attribute__((ext_vector_type(16))) unsigned short v16us;
typedef __attribute__((ext_vector_type(8)))  unsigned short v8us;
typedef __attribute__((ext_vector_type(8)))  unsigned       v8u;
typedef __attribute__((ext_vector_type(8)))  float          v8f;
typedef __attribute__((ext_vector_type(4)))  float          v4f;

#define CIN   128
#define COUT  256
#define HW    112
#define IMG   (HW * HW)        // 12544 pixels per image (32 | 12544)
#define PH    114              // padded input height/width
#define PIMG  (PH * PH)        // 12996
#define KG    (CIN * 9)        // 1152 = GEMM K
#define KSTEP 32
#define NSTEPS (KG / KSTEP)    // 36

#define WS_W_OFF   0                               // bf16 weights [256][1152]
#define WS_IN_OFF  (1u << 20)                      // padded bf16 input
#define WS_NEED    ((size_t)WS_IN_OFF + (size_t)16 * CIN * PIMG * 2)

// Two fp32 -> one packed bf16 pair (RNE) using v_perm_b32 for the high halves.
__device__ __forceinline__ unsigned pack2_bf16(float x, float y) {
  union { float f; unsigned u; } a, b;
  a.f = x; b.f = y;
  const unsigned ua = a.u + 0x7FFFu + ((a.u >> 16) & 1u);
  const unsigned ub = b.u + 0x7FFFu + ((b.u >> 16) & 1u);
  return __builtin_amdgcn_perm(ub, ua, 0x07060302u);
}

// ---------------------------------------------------------------- prep passes
// Weights fp32 OIHW -> bf16 [256][1152]. fp32 source is read-once -> NT loads;
// bf16 result is re-read by the main kernel -> regular (L2-resident) stores.
__global__ __launch_bounds__(256) void prep_weights(
    const float* __restrict__ wt, unsigned short* __restrict__ o) {
  const int t = blockIdx.x * 256 + threadIdx.x;       // 147456 pairs
  const float x = __builtin_nontemporal_load(wt + 2 * t);
  const float y = __builtin_nontemporal_load(wt + 2 * t + 1);
  ((unsigned*)o)[t] = pack2_bf16(x, y);
}

// Input fp32 NCHW -> zero-padded bf16 [16][128][114][114].
__global__ __launch_bounds__(256) void prep_input(
    const float* __restrict__ in, unsigned short* __restrict__ o) {
  const int t  = blockIdx.x * 256 + threadIdx.x;      // pair index, 57 per row
  const int pw2 = (t % 57) * 2;
  const int r   = t / 57;                             // row over [16*128][114]
  const int ph  = r % 114;
  const int rc  = r / 114;                            // n*128 + ci
  float x = 0.0f, y = 0.0f;
  if (ph >= 1 && ph <= HW) {
    const float* __restrict__ src = in + ((size_t)rc * HW + (ph - 1)) * HW;
    if (pw2 >= 1 && pw2 <= HW)         x = __builtin_nontemporal_load(src + pw2 - 1);
    if (pw2 + 1 >= 1 && pw2 + 1 <= HW) y = __builtin_nontemporal_load(src + pw2);
  }
  ((unsigned*)o)[t] = pack2_bf16(x, y);
}

// ------------------------------------------------------------ fragment loads
// A-fragment from pre-converted bf16 weights: two 16B loads, zero VALU.
__device__ __forceinline__ v16bf load_a_bf16(const unsigned short* __restrict__ row,
                                             int kbase, int ao) {
  const v8us lo = *(const v8us*)(row + kbase + ao);
  const v8us hi = *(const v8us*)(row + kbase + ao + 16);
  v16us w;
#pragma unroll
  for (int q = 0; q < 8; ++q) { w[q] = lo[q]; w[8 + q] = hi[q]; }
  return __builtin_bit_cast(v16bf, w);
}

// A-fragment from fp32 weights with on-the-fly RNE convert (fallback path).
__device__ __forceinline__ v16bf load_a_f32(const float* __restrict__ row,
                                            int kbase, int ao) {
  const v4f a0 = *(const v4f*)(row + kbase + ao);
  const v4f a1 = *(const v4f*)(row + kbase + ao + 4);
  const v4f a2 = *(const v4f*)(row + kbase + ao + 16);
  const v4f a3 = *(const v4f*)(row + kbase + ao + 20);
  v8u u;
  u[0] = pack2_bf16(a0[0], a0[1]); u[1] = pack2_bf16(a0[2], a0[3]);
  u[2] = pack2_bf16(a1[0], a1[1]); u[3] = pack2_bf16(a1[2], a1[3]);
  u[4] = pack2_bf16(a2[0], a2[1]); u[5] = pack2_bf16(a2[2], a2[3]);
  u[6] = pack2_bf16(a3[0], a3[1]); u[7] = pack2_bf16(a3[2], a3[3]);
  return __builtin_bit_cast(v16bf, u);
}

// ---------------------------------------------------------------- B staging
// Fast path: 4 consecutive-k elements from padded bf16 input. No bounds
// checks, no converts: 4 u16 loads + 2 packs + one ds_store_b64.
__device__ __forceinline__ void stage4_pad(unsigned short* __restrict__ dstp,
                                           const unsigned short* __restrict__ inN,
                                           int& ci, int& kh, int& kw) {
  int c2 = ci, h2 = kh, w2 = kw;
  unsigned v[4];
#pragma unroll
  for (int c = 0; c < 4; ++c) {
    v[c] = inN[c2 * PIMG + h2 * PH + w2];
    ++w2;
    if (w2 == 3) { w2 = 0; ++h2; if (h2 == 3) { h2 = 0; ++c2; } }
  }
  *(unsigned*)(dstp)     = v[0] | (v[1] << 16);
  *(unsigned*)(dstp + 2) = v[2] | (v[3] << 16);
  ci += 3; kh += 1; kw += 2;                  // k += 32 = 3*9 + 5
  if (kw >= 3) { kw -= 3; ++kh; }
  if (kh >= 3) { kh -= 3; ++ci; }
}

// Fallback: fp32 gather with clamp+select (EXEC stays full) and RNE convert.
__device__ __forceinline__ void stage4_f32(unsigned short* __restrict__ dstp,
                                           const float* __restrict__ inN,
                                           int oh1, int ow1,
                                           int& ci, int& kh, int& kw) {
  int c2 = ci, h2 = kh, w2 = kw;
  float f[4];
#pragma unroll
  for (int c = 0; c < 4; ++c) {
    const int ih = oh1 + h2, iw = ow1 + w2;
    const bool ok = ((unsigned)ih < (unsigned)HW) & ((unsigned)iw < (unsigned)HW);
    const int ihc = ih < 0 ? 0 : (ih > HW - 1 ? HW - 1 : ih);
    const int iwc = iw < 0 ? 0 : (iw > HW - 1 ? HW - 1 : iw);
    const float v = inN[(c2 * HW + ihc) * HW + iwc];
    f[c] = ok ? v : 0.0f;
    ++w2;
    if (w2 == 3) { w2 = 0; ++h2; if (h2 == 3) { h2 = 0; ++c2; } }
  }
  *(unsigned*)(dstp)     = pack2_bf16(f[0], f[1]);
  *(unsigned*)(dstp + 2) = pack2_bf16(f[2], f[3]);
  ci += 3; kh += 1; kw += 2;
  if (kw >= 3) { kw -= 3; ++kh; }
  if (kh >= 3) { kh -= 3; ++ci; }
}

#define WMMA_BF16(A, B, C) \
  __builtin_amdgcn_wmma_f32_16x16x32_bf16(false, (A), false, (B), (short)0, (C), false, false)

// ------------------------------------------------------------- main kernels
// Implicit-GEMM conv. One block: all 256 Cout x 32 pixels; 8 waves, each wave
// 2 M-tiles x 2 N-tiles (4 WMMA / K-step). Double-buffered LDS B tile in
// WMMA B-fragment layout: [buf][ntile(2)][lane(32)][elem(16)] bf16.
__global__ __launch_bounds__(256) void conv3x3_wmma_fast(
    const unsigned short* __restrict__ inP,   // padded bf16 [16][128][114][114]
    const unsigned short* __restrict__ wtb,   // bf16 [256][1152]
    const float* __restrict__ bias,
    float* __restrict__ out)
{
  __shared__ __align__(32) unsigned short Bsh[2][2 * 32 * 16];

  const int tid  = threadIdx.x;
  const int lane = tid & 31;
  const int wave = tid >> 5;

  const int pixBase = blockIdx.x * 32;
  const int nImg    = pixBase / IMG;
  const int piBase  = pixBase - nImg * IMG;

  const int p  = tid & 31;                    // pixel column
  const int j  = tid >> 5;                    // kk = j*4 + c
  const int pi = piBase + p;
  const int oh = pi / HW;
  const int ow = pi - oh * HW;
  // base at (ci=0,kh=0,kw=0): padded row = oh + kh, col = ow + kw
  const unsigned short* __restrict__ inN =
      inP + (size_t)nImg * CIN * PIMG + oh * PH + ow;

  int ci = (j * 4) / 9;
  int rr = j * 4 - ci * 9;
  int kh = rr / 3;
  int kw = rr - kh * 3;

  const int bl  = (p & 15) + ((j & 2) << 3);
  const int dst = ((p >> 4) << 9) + (bl << 4) + ((j & 4) << 1) + ((j & 1) << 2);

  const int am = wave * 32 + (lane & 15);
  const unsigned short* __restrict__ ar0 = wtb + (size_t)am * KG;
  const unsigned short* __restrict__ ar1 = ar0 + 16 * KG;
  const int ao = (lane & 16) ? 8 : 0;

  v8f acc00 = {}, acc01 = {}, acc10 = {}, acc11 = {};

  stage4_pad(&Bsh[0][dst], inN, ci, kh, kw);

  for (int ks = 0; ks < NSTEPS; ++ks) {
    const int kbase = ks * KSTEP;
    __syncthreads();

    const unsigned short* __restrict__ Bc = Bsh[ks & 1];
    const v16bf a0 = load_a_bf16(ar0, kbase, ao);
    const v16bf a1 = load_a_bf16(ar1, kbase, ao);
    const v16bf b0 = *(const v16bf*)(Bc + (lane << 4));
    const v16bf b1 = *(const v16bf*)(Bc + 512 + (lane << 4));

    if (ks + 1 < NSTEPS) {                    // uniform branch
      stage4_pad(&Bsh[(ks + 1) & 1][dst], inN, ci, kh, kw);
      __builtin_prefetch(ar0 + kbase + 2 * KSTEP, 0, 3);   // near-cache prefetch
      __builtin_prefetch(ar1 + kbase + 2 * KSTEP, 0, 3);
    }

    acc00 = WMMA_BF16(a0, b0, acc00);
    acc01 = WMMA_BF16(a0, b1, acc01);
    acc10 = WMMA_BF16(a1, b0, acc10);
    acc11 = WMMA_BF16(a1, b1, acc11);
  }

  // Streaming epilogue: written once, never re-read -> non-temporal stores so
  // the 196 MB output stream does not evict the L2-resident input/weights.
  const int mb0  = wave * 32 + ((lane & 16) ? 8 : 0);
  const int pcol = lane & 15;
  const long long outN = (long long)nImg * COUT * IMG + piBase + pcol;
#pragma unroll
  for (int r = 0; r < 8; ++r) {
    const int co0 = mb0 + r;
    const int co1 = co0 + 16;
    const float bv0 = bias[co0];
    const float bv1 = bias[co1];
    const long long o0 = outN + (long long)co0 * IMG;
    const long long o1 = outN + (long long)co1 * IMG;
    __builtin_nontemporal_store(acc00[r] + bv0, out + o0);
    __builtin_nontemporal_store(acc01[r] + bv0, out + o0 + 16);
    __builtin_nontemporal_store(acc10[r] + bv1, out + o1);
    __builtin_nontemporal_store(acc11[r] + bv1, out + o1 + 16);
  }
}

// Fallback (scratch too small): identical structure, fp32 sources on the fly.
__global__ __launch_bounds__(256) void conv3x3_wmma_fb(
    const float* __restrict__ in, const float* __restrict__ wt,
    const float* __restrict__ bias, float* __restrict__ out)
{
  __shared__ __align__(32) unsigned short Bsh[2][2 * 32 * 16];

  const int tid  = threadIdx.x;
  const int lane = tid & 31;
  const int wave = tid >> 5;

  const int pixBase = blockIdx.x * 32;
  const int nImg    = pixBase / IMG;
  const int piBase  = pixBase - nImg * IMG;

  const int p  = tid & 31;
  const int j  = tid >> 5;
  const int pi = piBase + p;
  const int oh1 = pi / HW - 1;
  const int ow1 = pi - (pi / HW) * HW - 1;
  const float* __restrict__ inN = in + (long long)nImg * CIN * IMG;

  int ci = (j * 4) / 9;
  int rr = j * 4 - ci * 9;
  int kh = rr / 3;
  int kw = rr - kh * 3;

  const int bl  = (p & 15) + ((j & 2) << 3);
  const int dst = ((p >> 4) << 9) + (bl << 4) + ((j & 4) << 1) + ((j & 1) << 2);

  const int am = wave * 32 + (lane & 15);
  const float* __restrict__ ar0 = wt + (long long)am * KG;
  const float* __restrict__ ar1 = ar0 + 16 * KG;
  const int ao = (lane & 16) ? 8 : 0;

  v8f acc00 = {}, acc01 = {}, acc10 = {}, acc11 = {};

  stage4_f32(&Bsh[0][dst], inN, oh1, ow1, ci, kh, kw);

  for (int ks = 0; ks < NSTEPS; ++ks) {
    const int kbase = ks * KSTEP;
    __syncthreads();

    const unsigned short* __restrict__ Bc = Bsh[ks & 1];
    const v16bf a0 = load_a_f32(ar0, kbase, ao);
    const v16bf a1 = load_a_f32(ar1, kbase, ao);
    const v16bf b0 = *(const v16bf*)(Bc + (lane << 4));
    const v16bf b1 = *(const v16bf*)(Bc + 512 + (lane << 4));

    if (ks + 1 < NSTEPS) {
      stage4_f32(&Bsh[(ks + 1) & 1][dst], inN, oh1, ow1, ci, kh, kw);
      __builtin_prefetch(ar0 + kbase + 2 * KSTEP, 0, 3);
      __builtin_prefetch(ar1 + kbase + 2 * KSTEP, 0, 3);
    }

    acc00 = WMMA_BF16(a0, b0, acc00);
    acc01 = WMMA_BF16(a0, b1, acc01);
    acc10 = WMMA_BF16(a1, b0, acc10);
    acc11 = WMMA_BF16(a1, b1, acc11);
  }

  const int mb0  = wave * 32 + ((lane & 16) ? 8 : 0);
  const int pcol = lane & 15;
  const long long outN = (long long)nImg * COUT * IMG + piBase + pcol;
#pragma unroll
  for (int r = 0; r < 8; ++r) {
    const int co0 = mb0 + r;
    const int co1 = co0 + 16;
    const float bv0 = bias[co0];
    const float bv1 = bias[co1];
    const long long o0 = outN + (long long)co0 * IMG;
    const long long o1 = outN + (long long)co1 * IMG;
    __builtin_nontemporal_store(acc00[r] + bv0, out + o0);
    __builtin_nontemporal_store(acc01[r] + bv0, out + o0 + 16);
    __builtin_nontemporal_store(acc10[r] + bv1, out + o1);
    __builtin_nontemporal_store(acc11[r] + bv1, out + o1 + 16);
  }
}

// Extra tuple outputs: patch_location_tensor (i32 bits) + out_p_h = out_p_w = 10.
__global__ void tail_outputs(const int* __restrict__ patch, float* __restrict__ out) {
  const long long base = (long long)16 * COUT * IMG;
  int* o = (int*)(out + base);
  const int i = threadIdx.x;
  if (i < 32) o[i] = patch[i];
  if (i == 32) o[32] = 10;
  if (i == 33) o[33] = 10;
}

extern "C" void kernel_launch(void* const* d_in, const int* in_sizes, int n_in,
                              void* d_out, int out_size, void* d_ws, size_t ws_size,
                              hipStream_t stream) {
  (void)in_sizes; (void)n_in; (void)out_size;
  const float* in    = (const float*)d_in[0];
  const float* wt    = (const float*)d_in[1];
  const float* bias  = (const float*)d_in[2];
  const int*   patch = (const int*)d_in[4];
  float* out = (float*)d_out;

  if (ws_size >= WS_NEED) {
    unsigned short* wbf  = (unsigned short*)((char*)d_ws + WS_W_OFF);
    unsigned short* ipad = (unsigned short*)((char*)d_ws + WS_IN_OFF);
    prep_weights<<<dim3((COUT * KG / 2) / 256), 256, 0, stream>>>(wt, wbf);
    prep_input<<<dim3((16 * CIN * PH * 57) / 256), 256, 0, stream>>>(in, ipad);
    conv3x3_wmma_fast<<<dim3((16 * IMG) / 32), 256, 0, stream>>>(ipad, wbf, bias, out);
  } else {
    conv3x3_wmma_fb<<<dim3((16 * IMG) / 32), 256, 0, stream>>>(in, wt, bias, out);
  }
  tail_outputs<<<1, 64, 0, stream>>>(patch, out);
}
